// RotarySelfAttention_20074677142373
// MI455X (gfx1250) — matrix operand order repaired and checked by reading, not compiled, not run on hardware
//
#include <hip/hip_runtime.h>
#include <hip/hip_bf16.h>

// ---------------------------------------------------------------------------
// Rotary self-attention forward for MI455X (gfx1250), wave32 + WMMA bf16.
// B=2, S=2048, E=1024, H=16, DH=64.
// Pipeline: f32->bf16 pre-convert; QKV GEMM; RoPE; flash-attention (async K
// staging via GLOBAL_LOAD_ASYNC_TO_LDS_B128); out GEMM.
// ---------------------------------------------------------------------------

typedef __bf16 bf16;
typedef __attribute__((ext_vector_type(16))) __bf16 v16bf;
typedef __attribute__((ext_vector_type(8)))  __bf16 v8bf;
typedef __attribute__((ext_vector_type(8)))  float  v8f;

#define WMMA_BF16(a, b, c) \
    __builtin_amdgcn_wmma_f32_16x16x32_bf16(false, (a), false, (b), (short)0, (c), false, false)

static constexpr int Bc = 2;
static constexpr int Sc = 2048;
static constexpr int Ec = 1024;
static constexpr int Hc = 16;
static constexpr int DHc = 64;

// ---- fragment load helpers (A/B use contiguous k order, same for both) ------
// A (16x32, bf16): lane L: m = L%16, hi = L/16; element e -> K = hi*16 + e
// B (32x16, bf16): lane L: n = L%16, hi = L/16; element e -> K = hi*16 + e
// C (16x16, f32) : lane L: n = L%16;            vgpr r    -> M = r + 8*(L/16)

__device__ __forceinline__ v16bf load_a_bf16(const bf16* __restrict__ row, int k0, int hi) {
    const v8bf lo = *(const v8bf*)(row + k0 + hi * 16);
    const v8bf hv = *(const v8bf*)(row + k0 + hi * 16 + 8);
    v16bf a;
#pragma unroll
    for (int i = 0; i < 8; ++i) { a[i] = lo[i]; a[8 + i] = hv[i]; }
    return a;
}

__device__ __forceinline__ v16bf load_b_bf16(const bf16* __restrict__ row, int ks) {
    return *(const v16bf*)(row + ks);
}

// Async 16-byte global -> LDS copy (per lane), tracked by ASYNCcnt.
__device__ __forceinline__ void async_copy_b128(void* lds_dst, const void* gsrc) {
    const unsigned int  la = (unsigned int)(unsigned long long)lds_dst; // addr[31:0] = LDS offset
    const unsigned long long ga = (unsigned long long)gsrc;
    asm volatile("global_load_async_to_lds_b128 %0, %1, off"
                 :: "v"(la), "v"(ga) : "memory");
}

__device__ __forceinline__ void wait_asynccnt0() {
    asm volatile("s_wait_asynccnt 0x0" ::: "memory");
}

// ---------------------------------------------------------------------------
// Kernel 0: fp32 -> bf16 conversion (x and weights), 4 elems/thread.
// ---------------------------------------------------------------------------
__global__ __launch_bounds__(256) void cvt_bf16_kernel(
    const float* __restrict__ src, bf16* __restrict__ dst) {
    const size_t i = ((size_t)blockIdx.x * blockDim.x + threadIdx.x) * 4;
    const float4 f = *(const float4*)(src + i);
    dst[i + 0] = (bf16)f.x;
    dst[i + 1] = (bf16)f.y;
    dst[i + 2] = (bf16)f.z;
    dst[i + 3] = (bf16)f.w;
}

// ---------------------------------------------------------------------------
// Kernel 1: fused QKV projection. out = x @ W.T reshaped to [B,H,S,DH] bf16.
// grid = (BS/128, E/128, 3); block = 256 (8 waves), wave -> 32M x 64N tile.
// ---------------------------------------------------------------------------
__global__ __launch_bounds__(256) void qkv_proj_kernel(
    const bf16* __restrict__ xb,
    const bf16* __restrict__ Wq, const bf16* __restrict__ Wk, const bf16* __restrict__ Wv,
    bf16* __restrict__ Qo, bf16* __restrict__ Ko, bf16* __restrict__ Vo) {

    const bf16* W = (blockIdx.z == 0) ? Wq : ((blockIdx.z == 1) ? Wk : Wv);
    bf16* dst     = (blockIdx.z == 0) ? Qo : ((blockIdx.z == 1) ? Ko : Vo);

    const int lane = threadIdx.x & 31;
    const int wave = threadIdx.x >> 5;
    const int m0   = blockIdx.x * 128 + (wave >> 1) * 32;  // 4 M-groups
    const int n0   = blockIdx.y * 128 + (wave & 1) * 64;   // 2 N-groups
    const int mrow = lane & 15;
    const int hi   = lane >> 4;

    v8f c[2][4];
#pragma unroll
    for (int g = 0; g < 2; ++g)
#pragma unroll
        for (int t = 0; t < 4; ++t)
#pragma unroll
            for (int r = 0; r < 8; ++r) c[g][t][r] = 0.0f;

    const bf16* arow0 = xb + (size_t)(m0 + mrow) * Ec;
    const bf16* arow1 = xb + (size_t)(m0 + 16 + mrow) * Ec;
    for (int k0 = 0; k0 < Ec; k0 += 32) {
        const v16bf a0 = load_a_bf16(arow0, k0, hi);
        const v16bf a1 = load_a_bf16(arow1, k0, hi);
        const int ks = k0 + hi * 16;
#pragma unroll
        for (int t = 0; t < 4; ++t) {
            const v16bf bm = load_b_bf16(W + (size_t)(n0 + t * 16 + mrow) * Ec, ks);
            c[0][t] = WMMA_BF16(a0, bm, c[0][t]);
            c[1][t] = WMMA_BF16(a1, bm, c[1][t]);
        }
    }

    // store with [B*S, E] -> [B, H, S, DH] transform
#pragma unroll
    for (int g = 0; g < 2; ++g)
#pragma unroll
        for (int t = 0; t < 4; ++t) {
            const int n = n0 + t * 16 + mrow;
            const int h = n >> 6, d = n & 63;
#pragma unroll
            for (int r = 0; r < 8; ++r) {
                const int m = m0 + g * 16 + r + 8 * hi;
                const int b = m >> 11, s = m & (Sc - 1);
                dst[(((size_t)b * Hc + h) * Sc + s) * DHc + d] = (bf16)c[g][t][r];
            }
        }
}

// ---------------------------------------------------------------------------
// Kernel 2: RoPE on Q and K (interleaved pairs, per reference).
// ---------------------------------------------------------------------------
__global__ __launch_bounds__(256) void rope_kernel(
    bf16* __restrict__ Q, bf16* __restrict__ K,
    const float* __restrict__ fcos, const float* __restrict__ fsin) {
    const size_t i = (size_t)blockIdx.x * blockDim.x + threadIdx.x; // B*H*S*(DH/2)
    const int d2 = (int)(i & 31);
    const size_t t = i >> 5;
    const int s = (int)(t & (Sc - 1));
    const size_t bh = t >> 11;
    const size_t base = (bh * Sc + s) * DHc + (size_t)d2 * 2;
    const float cs = fcos[(size_t)s * 32 + d2];
    const float sn = fsin[(size_t)s * 32 + d2];

    float qr = (float)Q[base], qi = (float)Q[base + 1];
    Q[base]     = (bf16)(qr * cs - qi * sn);
    Q[base + 1] = (bf16)(qr * sn + qi * cs);
    float kr = (float)K[base], ki = (float)K[base + 1];
    K[base]     = (bf16)(kr * cs - ki * sn);
    K[base + 1] = (bf16)(kr * sn + ki * cs);
}

// ---------------------------------------------------------------------------
// Kernel 3: flash attention per (b,h). grid = (B*H, S/128), block = 256.
// Wave owns a 16-query tile; keys streamed in chunks of 32 through LDS.
// K chunk staged via async global->LDS DMA; V staged transposed via VGPRs.
// O written as [B*S, E] bf16 (head-major columns) for the output projection.
// ---------------------------------------------------------------------------
__global__ __launch_bounds__(256) void attn_kernel(
    const bf16* __restrict__ Q, const bf16* __restrict__ K, const bf16* __restrict__ V,
    const float* __restrict__ mask, bf16* __restrict__ O) {

    __shared__ bf16 Kc[32][64];      // key chunk,   row-major  (4 KB)
    __shared__ bf16 Vt[64][32];      // value chunk, transposed (4 KB)
    __shared__ bf16 P[8][16][32];    // per-wave probability tile (8 KB)

    const int lane = threadIdx.x & 31;
    const int wave = threadIdx.x >> 5;
    const int bh   = blockIdx.x;            // b*H + h
    const int b    = bh >> 4;
    const int h    = bh & 15;
    const int m0   = blockIdx.y * 128 + wave * 16;
    const int mrow = lane & 15;
    const int hi   = lane >> 4;

    const bf16* Qb = Q + (size_t)bh * Sc * DHc;
    const bf16* Kb = K + (size_t)bh * Sc * DHc;
    const bf16* Vb = V + (size_t)bh * Sc * DHc;

    // Q A-fragments for d in [0,32) and [32,64) — loaded once, reused all chunks
    v16bf aQ0, aQ1;
    {
        const bf16* qrow = Qb + (size_t)(m0 + mrow) * DHc;
        const v8bf q0 = *(const v8bf*)(qrow + hi * 16);
        const v8bf q1 = *(const v8bf*)(qrow + 8 + hi * 16);
        const v8bf q2 = *(const v8bf*)(qrow + 32 + hi * 16);
        const v8bf q3 = *(const v8bf*)(qrow + 40 + hi * 16);
#pragma unroll
        for (int i = 0; i < 8; ++i) {
            aQ0[i] = q0[i]; aQ0[8 + i] = q1[i];
            aQ1[i] = q2[i]; aQ1[8 + i] = q3[i];
        }
    }

    float mrun[8], lrun[8];
    v8f acc[4];
#pragma unroll
    for (int t = 0; t < 4; ++t)
#pragma unroll
        for (int r = 0; r < 8; ++r) acc[t][r] = 0.0f;
#pragma unroll
    for (int r = 0; r < 8; ++r) { mrun[r] = -3.0e38f; lrun[r] = 0.0f; }

    const float scale = 0.125f;  // 1/sqrt(DH)

    for (int kc = 0; kc < Sc; kc += 32) {
        __syncthreads();
        {   // stage K chunk (async DMA) + V chunk transposed (via VGPRs)
            const int tid = threadIdx.x;
            const int row = tid >> 3;
            const int cb  = (tid & 7) * 8;
            const bf16* ks = Kb + (size_t)(kc + row) * DHc + cb;
            const bf16* vs = Vb + (size_t)(kc + row) * DHc + cb;
            async_copy_b128(&Kc[row][cb], ks);           // 16B/lane, ASYNCcnt
            const v8bf vv = *(const v8bf*)vs;
#pragma unroll
            for (int i = 0; i < 8; ++i) Vt[cb + i][row] = vv[i];
            // prefetch next chunk (global_prefetch_b8)
            if (kc + 32 < Sc) {
                __builtin_prefetch(ks + 32 * DHc, 0, 1);
                __builtin_prefetch(vs + 32 * DHc, 0, 1);
            }
        }
        wait_asynccnt0();
        __syncthreads();

        // scores for 32 keys: two 16-key C tiles, each = 2 WMMAs over DH=64
        v8f sc[2];
#pragma unroll
        for (int kt = 0; kt < 2; ++kt) {
            const bf16* krow = &Kc[kt * 16 + mrow][0];
            const v16bf b0 = *(const v16bf*)(krow + hi * 16);
            const v16bf b1 = *(const v16bf*)(krow + 32 + hi * 16);
            v8f c;
#pragma unroll
            for (int r = 0; r < 8; ++r) c[r] = 0.0f;
            c = WMMA_BF16(aQ0, b0, c);
            c = WMMA_BF16(aQ1, b1, c);
            const int kg = kc + kt * 16 + mrow;
#pragma unroll
            for (int r = 0; r < 8; ++r) {
                const int q = m0 + r + 8 * hi;
                c[r] = c[r] * scale + mask[(size_t)q * Sc + kg];
            }
            sc[kt] = c;
        }

        // online softmax update (row reductions live in one 16-lane half)
#pragma unroll
        for (int r = 0; r < 8; ++r) {
            float mx = fmaxf(sc[0][r], sc[1][r]);
#pragma unroll
            for (int off = 1; off < 16; off <<= 1)
                mx = fmaxf(mx, __shfl_xor(mx, off, 32));
            const float mn = fmaxf(mrun[r], mx);
            const float alpha = __expf(mrun[r] - mn);
            mrun[r] = mn;
            lrun[r] *= alpha;
#pragma unroll
            for (int t = 0; t < 4; ++t) acc[t][r] *= alpha;
            const float p0 = __expf(sc[0][r] - mn);
            const float p1 = __expf(sc[1][r] - mn);
            float ps = p0 + p1;
#pragma unroll
            for (int off = 1; off < 16; off <<= 1)
                ps += __shfl_xor(ps, off, 32);
            lrun[r] += ps;
            P[wave][r + 8 * hi][mrow]      = (bf16)p0;
            P[wave][r + 8 * hi][16 + mrow] = (bf16)p1;
        }

        // P (C-layout in LDS) -> A-fragment, then acc += P @ V
        v16bf aP;
        {
            const bf16* prow = &P[wave][mrow][0];
            const v8bf p0 = *(const v8bf*)(prow + hi * 16);
            const v8bf p1 = *(const v8bf*)(prow + 8 + hi * 16);
#pragma unroll
            for (int i = 0; i < 8; ++i) { aP[i] = p0[i]; aP[8 + i] = p1[i]; }
        }
#pragma unroll
        for (int t = 0; t < 4; ++t) {
            const v16bf bV = *(const v16bf*)(&Vt[t * 16 + mrow][hi * 16]);
            acc[t] = WMMA_BF16(aP, bV, acc[t]);
        }
    }

    // normalize and store O[b*S+q][h*64+d] as bf16
#pragma unroll
    for (int t = 0; t < 4; ++t) {
        const int d = t * 16 + mrow;
#pragma unroll
        for (int r = 0; r < 8; ++r) {
            const int q = m0 + r + 8 * hi;
            const float ov = acc[t][r] / lrun[r];
            O[((size_t)(b * Sc + q)) * Ec + h * 64 + d] = (bf16)ov;
        }
    }
}

// ---------------------------------------------------------------------------
// Kernel 4: output projection: out = O @ Wo.T + bo   (fp32 out)
// grid = (BS/128, E/128); wave -> 32M x 64N tile.
// ---------------------------------------------------------------------------
__global__ __launch_bounds__(256) void out_proj_kernel(
    const bf16* __restrict__ A, const bf16* __restrict__ Wo,
    const float* __restrict__ bo, float* __restrict__ out) {

    const int lane = threadIdx.x & 31;
    const int wave = threadIdx.x >> 5;
    const int m0   = blockIdx.x * 128 + (wave >> 1) * 32;
    const int n0   = blockIdx.y * 128 + (wave & 1) * 64;
    const int mrow = lane & 15;
    const int hi   = lane >> 4;

    v8f c[2][4];
#pragma unroll
    for (int g = 0; g < 2; ++g)
#pragma unroll
        for (int t = 0; t < 4; ++t)
#pragma unroll
            for (int r = 0; r < 8; ++r) c[g][t][r] = 0.0f;

    const bf16* arow0 = A + (size_t)(m0 + mrow) * Ec;
    const bf16* arow1 = A + (size_t)(m0 + 16 + mrow) * Ec;
    for (int k0 = 0; k0 < Ec; k0 += 32) {
        const v16bf a0 = load_a_bf16(arow0, k0, hi);
        const v16bf a1 = load_a_bf16(arow1, k0, hi);
        const int ks = k0 + hi * 16;
#pragma unroll
        for (int t = 0; t < 4; ++t) {
            const v16bf bm = load_b_bf16(Wo + (size_t)(n0 + t * 16 + mrow) * Ec, ks);
            c[0][t] = WMMA_BF16(a0, bm, c[0][t]);
            c[1][t] = WMMA_BF16(a1, bm, c[1][t]);
        }
    }

#pragma unroll
    for (int g = 0; g < 2; ++g)
#pragma unroll
        for (int t = 0; t < 4; ++t) {
            const int n = n0 + t * 16 + mrow;
            const float bias = bo[n];
#pragma unroll
            for (int r = 0; r < 8; ++r) {
                const int m = m0 + g * 16 + r + 8 * hi;
                out[(size_t)m * Ec + n] = c[g][t][r] + bias;
            }
        }
}

// ---------------------------------------------------------------------------
extern "C" void kernel_launch(void* const* d_in, const int* in_sizes, int n_in,
                              void* d_out, int out_size, void* d_ws, size_t ws_size,
                              hipStream_t stream) {
    (void)in_sizes; (void)n_in; (void)out_size; (void)ws_size;

    const float* x    = (const float*)d_in[0];
    const float* Wq   = (const float*)d_in[1];
    const float* Wk   = (const float*)d_in[2];
    const float* Wv   = (const float*)d_in[3];
    const float* Wo   = (const float*)d_in[4];
    const float* bo   = (const float*)d_in[5];
    const float* fcos = (const float*)d_in[6];
    const float* fsin = (const float*)d_in[7];
    const float* mask = (const float*)d_in[8];

    const size_t qkv_elems = (size_t)Bc * Hc * Sc * DHc;   // 4,194,304
    const size_t w_elems   = (size_t)Ec * Ec;              // 1,048,576
    bf16* Qw  = (bf16*)d_ws;
    bf16* Kw  = Qw + qkv_elems;
    bf16* Vw  = Kw + qkv_elems;
    bf16* Ow  = Vw + qkv_elems;        // [B*S, E]
    bf16* xbw = Ow + qkv_elems;        // [B*S, E]
    bf16* Wqb = xbw + qkv_elems;
    bf16* Wkb = Wqb + w_elems;
    bf16* Wvb = Wkb + w_elems;
    bf16* Wob = Wvb + w_elems;

    dim3 blk(256);

    // fp32 -> bf16 pre-conversion (one-shot, bandwidth-trivial)
    cvt_bf16_kernel<<<(unsigned)(qkv_elems / 1024), blk, 0, stream>>>(x, xbw);
    cvt_bf16_kernel<<<(unsigned)(w_elems / 1024), blk, 0, stream>>>(Wq, Wqb);
    cvt_bf16_kernel<<<(unsigned)(w_elems / 1024), blk, 0, stream>>>(Wk, Wkb);
    cvt_bf16_kernel<<<(unsigned)(w_elems / 1024), blk, 0, stream>>>(Wv, Wvb);
    cvt_bf16_kernel<<<(unsigned)(w_elems / 1024), blk, 0, stream>>>(Wo, Wob);

    dim3 g1((Bc * Sc) / 128, Ec / 128, 3);                  // (32,8,3)
    qkv_proj_kernel<<<g1, blk, 0, stream>>>(xbw, Wqb, Wkb, Wvb, Qw, Kw, Vw);

    const size_t rope_threads = (size_t)Bc * Hc * Sc * (DHc / 2);
    rope_kernel<<<(unsigned)(rope_threads / 256), blk, 0, stream>>>(Qw, Kw, fcos, fsin);

    dim3 g2(Bc * Hc, Sc / 128);                             // (32,16)
    attn_kernel<<<g2, blk, 0, stream>>>(Qw, Kw, Vw, mask, Ow);

    dim3 g3((Bc * Sc) / 128, Ec / 128);                     // (32,8)
    out_proj_kernel<<<g3, blk, 0, stream>>>(Ow, Wob, bo, (float*)d_out);
}
